// Vim_74294344286651
// MI455X (gfx1250) — matrix operand order repaired and checked by reading, not compiled, not run on hardware
//
#include <hip/hip_runtime.h>
#include <hip/hip_bf16.h>
#include <math.h>

// ---- problem constants (match reference) ----
#define DIM     256
#define DT_RANK 32
#define D_STATE 16
#define DEPTH   12
#define BATCH   8
#define HH      32
#define WW      32
#define LSEQ    (HH * WW)        // 1024 tokens per image
#define MTOK    (BATCH * LSEQ)   // 8192 tokens total
#define LN_EPS  1e-5f

typedef __attribute__((ext_vector_type(16))) __bf16 v16bf;
typedef __attribute__((ext_vector_type(8)))  __bf16 v8bf;
typedef __attribute__((ext_vector_type(8)))  float  v8f;

// ---------------------------------------------------------------------------
// Coalesced transposes between (B, C, L) and (B, L, C)
// ---------------------------------------------------------------------------
__global__ void transpose_cl_to_lc(const float* __restrict__ in, float* __restrict__ out) {
    __shared__ float tile[32][33];
    int b  = blockIdx.z;
    int c0 = blockIdx.x * 32;
    int l0 = blockIdx.y * 32;
    int tx = threadIdx.x, ty = threadIdx.y;   // block (32, 8)
    #pragma unroll
    for (int i = ty; i < 32; i += 8)
        tile[i][tx] = in[((size_t)b * DIM + (c0 + i)) * LSEQ + l0 + tx];
    __syncthreads();
    #pragma unroll
    for (int i = ty; i < 32; i += 8)
        out[((size_t)b * LSEQ + (l0 + i)) * DIM + c0 + tx] = tile[tx][i];
}

__global__ void transpose_lc_to_cl(const float* __restrict__ in, float* __restrict__ out) {
    __shared__ float tile[32][33];
    int b  = blockIdx.z;
    int c0 = blockIdx.x * 32;
    int l0 = blockIdx.y * 32;
    int tx = threadIdx.x, ty = threadIdx.y;   // block (32, 8)
    #pragma unroll
    for (int i = ty; i < 32; i += 8)
        tile[i][tx] = in[((size_t)b * LSEQ + (l0 + i)) * DIM + c0 + tx];
    __syncthreads();
    #pragma unroll
    for (int i = ty; i < 32; i += 8)
        out[((size_t)b * DIM + (c0 + i)) * LSEQ + l0 + tx] = tile[tx][i];
}

// ---------------------------------------------------------------------------
// Bulk f32 -> bf16 conversion (weights, once per launch). n % 8 == 0.
// ---------------------------------------------------------------------------
__global__ void cvt_bf16_kernel(const float* __restrict__ in, __bf16* __restrict__ out, int n) {
    int i = ((int)blockIdx.x * (int)blockDim.x + (int)threadIdx.x) * 8;
    if (i >= n) return;
    float4 a = *(const float4*)(in + i);
    float4 b = *(const float4*)(in + i + 4);
    v8bf o;
    o[0] = (__bf16)a.x; o[1] = (__bf16)a.y; o[2] = (__bf16)a.z; o[3] = (__bf16)a.w;
    o[4] = (__bf16)b.x; o[5] = (__bf16)b.y; o[6] = (__bf16)b.z; o[7] = (__bf16)b.w;
    *(v8bf*)(out + i) = o;
}

// ---------------------------------------------------------------------------
// LayerNorm over channels: one wave32 per token, 8 channels per lane.
// Output written directly as bf16 (only consumer is the WMMA GEMM A-operand).
// ---------------------------------------------------------------------------
__global__ void ln_kernel(const float* __restrict__ x,
                          const float* __restrict__ g,
                          const float* __restrict__ bta,
                          __bf16* __restrict__ out) {
    int wave  = threadIdx.x >> 5;
    int lane  = threadIdx.x & 31;
    int token = blockIdx.x * 8 + wave;
    const float* xt = x + (size_t)token * DIM;

    float v[8];
    float s = 0.f, ss = 0.f;
    #pragma unroll
    for (int i = 0; i < 8; ++i) {
        v[i] = xt[lane * 8 + i];
        s  += v[i];
        ss += v[i] * v[i];
    }
    #pragma unroll
    for (int off = 16; off; off >>= 1) {
        s  += __shfl_xor(s,  off, 32);
        ss += __shfl_xor(ss, off, 32);
    }
    float mu   = s * (1.f / DIM);
    float var  = ss * (1.f / DIM) - mu * mu;
    float rstd = rsqrtf(var + LN_EPS);

    v8bf o;
    #pragma unroll
    for (int i = 0; i < 8; ++i) {
        int c = lane * 8 + i;
        o[i] = (__bf16)((v[i] - mu) * rstd * g[c] + bta[c]);
    }
    *(v8bf*)(out + (size_t)token * DIM + lane * 8) = o;
}

// ---------------------------------------------------------------------------
// WMMA bf16 GEMM:  C[M,N] = act( A[M,K](lda, bf16) * W[N,K]^T(bf16) + bias[N] )
// Block = 256 threads = 8 waves, arranged 4(M) x 2(N); each wave owns a 16x32
// output tile (two 16x16 accumulators sharing one A fragment) -> block tile
// 64x64. K templated so the k-loop fully unrolls (16 straight-line v_wmma for
// K=256). Fragments use contiguous k = half*16 + 0..15 in both A and B (wave32, bf16, 16x16x32);
// all fragment loads are 16-byte b128 loads of pre-packed bf16.
// ACT: 0 = identity, 1 = softplus. Optional bf16 shadow copy of the output.
// ---------------------------------------------------------------------------
__device__ __forceinline__ float softplus_f(float x) {
    return (x > 20.f) ? x : log1pf(__expf(x));
}

__device__ __forceinline__ v16bf frag16(const __bf16* p0, const __bf16* p1) {
    v8bf lo = *(const v8bf*)p0;
    v8bf hi = *(const v8bf*)p1;
    return __builtin_shufflevector(lo, hi, 0, 1, 2, 3, 4, 5, 6, 7,
                                           8, 9, 10, 11, 12, 13, 14, 15);
}

template <int K, int ACT>
__global__ void gemm_wmma_kernel(const __bf16* __restrict__ A, int lda,
                                 const __bf16* __restrict__ W,     // (N, K) row-major
                                 const float* __restrict__ bias,   // nullptr ok
                                 float* __restrict__ C,
                                 __bf16* __restrict__ Cbf,         // nullptr ok
                                 int ldc) {
    int wave = threadIdx.x >> 5;
    int lane = threadIdx.x & 31;
    int wm = wave & 3;                       // 4 waves along M
    int wn = wave >> 2;                      // 2 waves along N (x2 subtiles)
    int tile_m  = blockIdx.x * 64 + wm * 16;
    int tile_n0 = blockIdx.y * 64 + wn * 32;
    int half = lane >> 4;                    // K-half selector
    int l16  = lane & 15;                    // row (A) / col (B,C)

    const __bf16* Arow  = A + (size_t)(tile_m + l16) * lda;
    const __bf16* Wrow0 = W + (size_t)(tile_n0 + l16) * K;
    const __bf16* Wrow1 = W + (size_t)(tile_n0 + 16 + l16) * K;

    v8f acc0 = {}, acc1 = {};
    #pragma unroll
    for (int k0 = 0; k0 < K; k0 += 32) {
        if (k0 + 32 < K) {                   // gfx1250 global_prefetch_b8
            __builtin_prefetch(Arow  + k0 + 32, 0, 1);
            __builtin_prefetch(Wrow0 + k0 + 32, 0, 1);
            __builtin_prefetch(Wrow1 + k0 + 32, 0, 1);
        }
        // A 16x32: lane m = l16; elements span K = half*16 + 0..15
        v16bf afrag = frag16(Arow + k0 + half * 16, Arow + k0 + half * 16 + 8);
        // B 32x16: lane col = l16; elements span K = half*16 + 0..15
        v16bf bfrag0 = frag16(Wrow0 + k0 + half * 16, Wrow0 + k0 + half * 16 + 8);
        v16bf bfrag1 = frag16(Wrow1 + k0 + half * 16, Wrow1 + k0 + half * 16 + 8);

        acc0 = __builtin_amdgcn_wmma_f32_16x16x32_bf16(
            false, afrag, false, bfrag0, (short)0, acc0, false, false);
        acc1 = __builtin_amdgcn_wmma_f32_16x16x32_bf16(
            false, afrag, false, bfrag1, (short)0, acc1, false, false);
    }

    // C 16x16 f32: lane col n = l16; VGPR v -> row m = half*8 + v
    int n0 = tile_n0 + l16;
    int n1 = n0 + 16;
    float b0 = bias ? bias[n0] : 0.f;
    float b1 = bias ? bias[n1] : 0.f;
    #pragma unroll
    for (int v = 0; v < 8; ++v) {
        int m = tile_m + half * 8 + v;
        float r0 = acc0[v] + b0;
        float r1 = acc1[v] + b1;
        if (ACT == 1) { r0 = softplus_f(r0); r1 = softplus_f(r1); }
        size_t base = (size_t)m * ldc;
        C[base + n0] = r0;
        C[base + n1] = r1;
        if (Cbf) {
            Cbf[base + n0] = (__bf16)r0;
            Cbf[base + n1] = (__bf16)r1;
        }
    }
}

// ---------------------------------------------------------------------------
// Selective scan. One thread per (batch, channel); 16 states in registers.
// reverse==1 walks t = L-1..0 (equivalent to reversing inputs+outputs, since
// every non-scan op is per-token).
// dbc layout per token: [0:32) delta_rank, [32:48) B, [48:64) C
// ---------------------------------------------------------------------------
__global__ void scan_kernel(const float* __restrict__ u,
                            const float* __restrict__ delta,
                            const float* __restrict__ dbc,
                            const float* __restrict__ A_log,  // (DIM, D_STATE)
                            const float* __restrict__ Dp,     // (DIM)
                            float* __restrict__ y,
                            int reverse) {
    int b = blockIdx.x;
    int d = blockIdx.y * 64 + threadIdx.x;

    float Avals[D_STATE];
    #pragma unroll
    for (int n = 0; n < D_STATE; ++n)
        Avals[n] = -__expf(A_log[d * D_STATE + n]);

    float h[D_STATE];
    #pragma unroll
    for (int n = 0; n < D_STATE; ++n) h[n] = 0.f;
    float Dd = Dp[d];

    for (int t = 0; t < LSEQ; ++t) {
        int tt = reverse ? (LSEQ - 1 - t) : t;
        size_t tok = (size_t)b * LSEQ + tt;
        float dlt = delta[tok * DIM + d];
        float ut  = u[tok * DIM + d];
        // B/C are uniform across the 64 lanes of this block -> broadcast loads
        const float4* bq = (const float4*)(dbc + tok * 64 + 32);
        float4 B0 = bq[0], B1 = bq[1], B2 = bq[2], B3 = bq[3];   // B[0..15]
        float4 C0 = bq[4], C1 = bq[5], C2 = bq[6], C3 = bq[7];   // C[0..15]
        float Bv[D_STATE] = {B0.x, B0.y, B0.z, B0.w, B1.x, B1.y, B1.z, B1.w,
                             B2.x, B2.y, B2.z, B2.w, B3.x, B3.y, B3.z, B3.w};
        float Cv[D_STATE] = {C0.x, C0.y, C0.z, C0.w, C1.x, C1.y, C1.z, C1.w,
                             C2.x, C2.y, C2.z, C2.w, C3.x, C3.y, C3.z, C3.w};
        float du  = dlt * ut;
        float acc = 0.f;
        #pragma unroll
        for (int n = 0; n < D_STATE; ++n) {
            float dA = __expf(dlt * Avals[n]);
            h[n] = dA * h[n] + du * Bv[n];
            acc += h[n] * Cv[n];
        }
        y[tok * DIM + d] = acc + ut * Dd;
    }
}

// ---------------------------------------------------------------------------
// seq = (y1 + y2) * silu(xp) + seq    (float4 vectorized)
// ---------------------------------------------------------------------------
__global__ void combine_kernel(const float* __restrict__ xp,
                               const float* __restrict__ y1,
                               const float* __restrict__ y2,
                               float* __restrict__ seq) {
    size_t i = ((size_t)blockIdx.x * blockDim.x + threadIdx.x) * 4;
    float4 p  = *(const float4*)(xp + i);
    float4 a  = *(const float4*)(y1 + i);
    float4 b  = *(const float4*)(y2 + i);
    float4 s  = *(const float4*)(seq + i);
    float4 r;
    r.x = (a.x + b.x) * (p.x / (1.f + __expf(-p.x))) + s.x;
    r.y = (a.y + b.y) * (p.y / (1.f + __expf(-p.y))) + s.y;
    r.z = (a.z + b.z) * (p.z / (1.f + __expf(-p.z))) + s.z;
    r.w = (a.w + b.w) * (p.w / (1.f + __expf(-p.w))) + s.w;
    *(float4*)(seq + i) = r;
}

// ---------------------------------------------------------------------------
extern "C" void kernel_launch(void* const* d_in, const int* in_sizes, int n_in,
                              void* d_out, int out_size, void* d_ws, size_t ws_size,
                              hipStream_t stream) {
    (void)in_sizes; (void)n_in; (void)out_size; (void)ws_size;

    const float* x       = (const float*)d_in[0];   // (B, d, H, W)
    const float* ln_g    = (const float*)d_in[1];   // (L, d)
    const float* ln_b    = (const float*)d_in[2];
    const float* proj_w  = (const float*)d_in[3];   // (L, d, d)
    const float* proj_b  = (const float*)d_in[4];
    const float* convf_w = (const float*)d_in[5];
    const float* convf_b = (const float*)d_in[6];
    const float* convb_w = (const float*)d_in[7];
    const float* convb_b = (const float*)d_in[8];
    const float* dbc_w   = (const float*)d_in[9];   // (L, 64, d)
    const float* dtp_w   = (const float*)d_in[10];  // (L, d, 32)
    const float* dtp_b   = (const float*)d_in[11];
    const float* A_log   = (const float*)d_in[12];  // (L, d, 16)
    const float* D_par   = (const float*)d_in[13];  // (L, d)
    float* out = (float*)d_out;

    // ---- workspace carve-up (everything fits in MI455X's 192 MB L2) ----
    char* ws = (char*)d_ws;
    const size_t SZ_ACT  = (size_t)MTOK * DIM * sizeof(float);   // 8 MB
    const size_t SZ_DBC  = (size_t)MTOK * 64  * sizeof(float);   // 2 MB
    const size_t SZ_ACTB = (size_t)MTOK * DIM * sizeof(__bf16);  // 4 MB
    const size_t SZ_DBCB = (size_t)MTOK * 64  * sizeof(__bf16);  // 1 MB
    float* seq   = (float*)ws; ws += SZ_ACT;   // residual stream (B,L,d)
    float* xp    = (float*)ws; ws += SZ_ACT;
    float* xf    = (float*)ws; ws += SZ_ACT;
    float* xbp   = (float*)ws; ws += SZ_ACT;
    float* dltf  = (float*)ws; ws += SZ_ACT;
    float* dltb  = (float*)ws; ws += SZ_ACT;
    float* y1    = (float*)ws; ws += SZ_ACT;
    float* y2    = (float*)ws; ws += SZ_ACT;
    float* dbcf  = (float*)ws; ws += SZ_DBC;
    float* dbcb  = (float*)ws; ws += SZ_DBC;
    __bf16* xn_bf   = (__bf16*)ws; ws += SZ_ACTB;
    __bf16* xp_bf   = (__bf16*)ws; ws += SZ_ACTB;
    __bf16* xf_bf   = (__bf16*)ws; ws += SZ_ACTB;
    __bf16* xbp_bf  = (__bf16*)ws; ws += SZ_ACTB;
    __bf16* dbcf_bf = (__bf16*)ws; ws += SZ_DBCB;
    __bf16* dbcb_bf = (__bf16*)ws; ws += SZ_DBCB;
    // bf16 weights for all layers (converted once per launch)
    const int N_PW  = DEPTH * DIM * DIM;       // 786432 each for proj/convf/convb
    const int N_DBC = DEPTH * 64 * DIM;        // 196608
    const int N_DTP = DEPTH * DIM * DT_RANK;   // 98304
    __bf16* pw_bf  = (__bf16*)ws; ws += (size_t)N_PW  * sizeof(__bf16);
    __bf16* cfw_bf = (__bf16*)ws; ws += (size_t)N_PW  * sizeof(__bf16);
    __bf16* cbw_bf = (__bf16*)ws; ws += (size_t)N_PW  * sizeof(__bf16);
    __bf16* dbw_bf = (__bf16*)ws; ws += (size_t)N_DBC * sizeof(__bf16);
    __bf16* dtw_bf = (__bf16*)ws; ws += (size_t)N_DTP * sizeof(__bf16);

    // ---- one-time weight conversion to bf16 ----
    cvt_bf16_kernel<<<N_PW  / 8 / 256, 256, 0, stream>>>(proj_w,  pw_bf,  N_PW);
    cvt_bf16_kernel<<<N_PW  / 8 / 256, 256, 0, stream>>>(convf_w, cfw_bf, N_PW);
    cvt_bf16_kernel<<<N_PW  / 8 / 256, 256, 0, stream>>>(convb_w, cbw_bf, N_PW);
    cvt_bf16_kernel<<<N_DBC / 8 / 256, 256, 0, stream>>>(dbc_w,   dbw_bf, N_DBC);
    cvt_bf16_kernel<<<N_DTP / 8 / 256, 256, 0, stream>>>(dtp_w,   dtw_bf, N_DTP);

    dim3 tb(32, 8);
    dim3 tg(DIM / 32, LSEQ / 32, BATCH);
    transpose_cl_to_lc<<<tg, tb, 0, stream>>>(x, seq);

    dim3 gemm_blk(256);
    dim3 g_dd(MTOK / 64, DIM / 64);     // 256-col GEMMs: 128 x 4 blocks
    dim3 g_dbc(MTOK / 64, 1);           // 64-col GEMM
    dim3 scan_grid(BATCH, DIM / 64);

    for (int layer = 0; layer < DEPTH; ++layer) {
        const float* g    = ln_g    + (size_t)layer * DIM;
        const float* bta  = ln_b    + (size_t)layer * DIM;
        const float* pb   = proj_b  + (size_t)layer * DIM;
        const float* cfb  = convf_b + (size_t)layer * DIM;
        const float* cbb  = convb_b + (size_t)layer * DIM;
        const float* dtpb = dtp_b   + (size_t)layer * DIM;
        const float* alog = A_log   + (size_t)layer * DIM * D_STATE;
        const float* dpl  = D_par   + (size_t)layer * DIM;
        const __bf16* pw   = pw_bf  + (size_t)layer * DIM * DIM;
        const __bf16* cfw  = cfw_bf + (size_t)layer * DIM * DIM;
        const __bf16* cbw  = cbw_bf + (size_t)layer * DIM * DIM;
        const __bf16* dbcw = dbw_bf + (size_t)layer * 64 * DIM;
        const __bf16* dtpw = dtw_bf + (size_t)layer * DIM * DT_RANK;

        // LayerNorm -> bf16
        ln_kernel<<<MTOK / 8, 256, 0, stream>>>(seq, g, bta, xn_bf);

        // xp = xn @ pw^T + pb      (f32 + bf16 shadow)
        gemm_wmma_kernel<DIM, 0><<<g_dd, gemm_blk, 0, stream>>>(
            xn_bf, DIM, pw, pb, xp, xp_bf, DIM);
        // xf  = softplus(xp @ cfw^T + cfb)
        gemm_wmma_kernel<DIM, 1><<<g_dd, gemm_blk, 0, stream>>>(
            xp_bf, DIM, cfw, cfb, xf, xf_bf, DIM);
        // xbp = softplus(xp @ cbw^T + cbb)   (reversal handled inside scan)
        gemm_wmma_kernel<DIM, 1><<<g_dd, gemm_blk, 0, stream>>>(
            xp_bf, DIM, cbw, cbb, xbp, xbp_bf, DIM);

        // ---- forward SSM ----
        gemm_wmma_kernel<DIM, 0><<<g_dbc, gemm_blk, 0, stream>>>(
            xf_bf, DIM, dbcw, nullptr, dbcf, dbcf_bf, 64);
        gemm_wmma_kernel<DT_RANK, 1><<<g_dd, gemm_blk, 0, stream>>>(
            dbcf_bf, 64, dtpw, dtpb, dltf, nullptr, DIM);
        scan_kernel<<<scan_grid, 64, 0, stream>>>(
            xf, dltf, dbcf, alog, dpl, y1, /*reverse=*/0);

        // ---- backward SSM ----
        gemm_wmma_kernel<DIM, 0><<<g_dbc, gemm_blk, 0, stream>>>(
            xbp_bf, DIM, dbcw, nullptr, dbcb, dbcb_bf, 64);
        gemm_wmma_kernel<DT_RANK, 1><<<g_dd, gemm_blk, 0, stream>>>(
            dbcb_bf, 64, dtpw, dtpb, dltb, nullptr, DIM);
        scan_kernel<<<scan_grid, 64, 0, stream>>>(
            xbp, dltb, dbcb, alog, dpl, y2, /*reverse=*/1);

        // residual update: seq = (y1 + y2) * silu(xp) + seq
        combine_kernel<<<(MTOK * DIM) / 4 / 256, 256, 0, stream>>>(xp, y1, y2, seq);
    }

    transpose_lc_to_cl<<<tg, tb, 0, stream>>>(seq, out);
}
